// Decoder_2353642078763
// MI455X (gfx1250) — matrix operand order repaired and checked by reading, not compiled, not run on hardware
//
#include <hip/hip_runtime.h>
#include <hip/hip_bf16.h>

// Problem constants
constexpr int kB = 32, kS = 128, kT = 96, kH = 1024, kV = 32000;

typedef __attribute__((ext_vector_type(16))) _Float16 v16h;
typedef __attribute__((ext_vector_type(8)))  _Float16 h8;
typedef __attribute__((ext_vector_type(8)))  float    v8f;
typedef __attribute__((ext_vector_type(4)))  float    f32x4;
typedef __attribute__((ext_vector_type(4)))  unsigned int u32x4;
typedef __attribute__((ext_vector_type(8)))  int      i32x8;
typedef __attribute__((ext_vector_type(4)))  int      i32x4;

__device__ inline v8f wmma_f16(v16h a, v16h b, v8f c) {
  // D = A(16x32 f16) * B(32x16 f16) + C(16x16 f32)
  return __builtin_amdgcn_wmma_f32_16x16x32_f16(false, a, false, b, (short)0, c,
                                                false, false);
}

__device__ inline int lane_id() { return threadIdx.x & 31; }

// ---- Tensor Data Mover: 2D f16 tile (tile_k contiguous elems x tile_rows)
// from global row-major (stride_elems) into LDS at byte offset lds_off.
// D# packing per CDNA5 ISA (count=1, type=2,
// data_size=1 -> 2 bytes). Groups 2/3/aux zero (2D tensor).
__device__ inline void tdm_load_2d_f16(unsigned lds_off, const void* gaddr,
                                       unsigned tile_k, unsigned tile_rows,
                                       unsigned stride_elems,
                                       unsigned tensor_d0, unsigned tensor_d1) {
  unsigned long long ga = (unsigned long long)(size_t)gaddr;
  u32x4 g0;
  g0[0] = 1u;                                    // count=1 (valid), user mode
  g0[1] = lds_off;                               // lds_addr (bytes)
  g0[2] = (unsigned)(ga & 0xffffffffu);          // global_addr[31:0]
  g0[3] = (unsigned)((ga >> 32) & 0x01ffffffu)   // global_addr[56:32]
          | (2u << 30);                          // type = 2 ("image")
  i32x8 g1;
  g1[0] = (int)(1u << 16);                       // data_size=1 (2B); wg_mask=0
  g1[1] = (int)((tensor_d0 & 0xffffu) << 16);    // tensor_dim0[15:0] @bits63:48
  g1[2] = (int)((tensor_d0 >> 16) | ((tensor_d1 & 0xffffu) << 16));
  g1[3] = (int)((tensor_d1 >> 16) | (tile_k << 16));  // tile_dim0 @bits127:112
  g1[4] = (int)(tile_rows & 0xffffu);            // tile_dim1 (tile_dim2 = 0)
  g1[5] = (int)stride_elems;                     // tensor_dim0_stride[31:0]
  g1[6] = 0;                                     // stride[47:32]=0, dim1_stride=0
  g1[7] = 0;
  i32x4 z4 = {0, 0, 0, 0};
  i32x8 z8 = {0, 0, 0, 0, 0, 0, 0, 0};
  __builtin_amdgcn_tensor_load_to_lds(g0, g1, z4, z4, z8, 0);
}

// ---- Fragment loaders (layouts per CDNA5 ISA 7.12.2) -----------------------

// A fragment (16x32, f16) from row-major f32 matrix, tile origin (m0,k0).
__device__ inline v16h load_a_f32(const float* __restrict__ A, long lda, int m0, int k0) {
  int l = lane_id();
  const float* p = A + (size_t)(m0 + (l & 15)) * lda + (k0 + ((l >> 4) << 4));
  const f32x4* q = reinterpret_cast<const f32x4*>(p);
  f32x4 x0 = q[0], x1 = q[1], y0 = q[2], y1 = q[3];
  v16h a;
#pragma unroll
  for (int i = 0; i < 4; ++i) {
    a[i]      = (_Float16)x0[i];
    a[i + 4]  = (_Float16)x1[i];
    a[i + 8]  = (_Float16)y0[i];
    a[i + 12] = (_Float16)y1[i];
  }
  return a;
}

// A fragment from row-major f16 matrix (two 128-bit loads; works for LDS too).
__device__ inline v16h load_a_f16(const _Float16* __restrict__ A, long lda, int m0, int k0) {
  int l = lane_id();
  const _Float16* p = A + (size_t)(m0 + (l & 15)) * lda + (k0 + ((l >> 4) << 3));
  const h8* q = reinterpret_cast<const h8*>(p);
  h8 lo = q[0], hi = q[2];
  v16h a;
#pragma unroll
  for (int i = 0; i < 8; ++i) { a[i] = lo[i]; a[i + 8] = hi[i]; }
  return a;
}

// B fragment (32x16, f16) from row-major f32 weight W[K][N], tile (k0,n0).
__device__ inline v16h load_b_f32(const float* __restrict__ W, long ldb, int k0, int n0) {
  int l = lane_id();
  const float* p = W + (size_t)(k0 + ((l >> 4) << 4)) * ldb + (n0 + (l & 15));
  v16h b;
#pragma unroll
  for (int i = 0; i < 16; ++i) b[i] = (_Float16)p[(size_t)i * ldb];
  return b;
}

// B fragment from LDS tile laid out [n][k], row stride 48 halves (16B aligned).
__device__ inline v16h load_b_lds(const _Float16* Bs, int nsub) {
  int l = lane_id();
  const _Float16* p = Bs + (size_t)(nsub + (l & 15)) * 48 + ((l >> 4) << 3);
  const h8* q = reinterpret_cast<const h8*>(p);
  h8 lo = q[0], hi = q[2];
  v16h b;
#pragma unroll
  for (int i = 0; i < 8; ++i) { b[i] = lo[i]; b[i + 8] = hi[i]; }
  return b;
}

// D store: n = n0 + lane%16 ; m = m0 + 8*(lane/16) + r
__device__ inline void store_d_bias(float* __restrict__ C, long ldc, int m0, int n0,
                                    v8f acc, const float* __restrict__ bias) {
  int l = lane_id();
  int n = n0 + (l & 15);
  int mb = m0 + ((l >> 4) << 3);
  float bv = bias ? bias[n] : 0.0f;
#pragma unroll
  for (int r = 0; r < 8; ++r) C[(size_t)(mb + r) * ldc + n] = acc[r] + bv;
}

// ---- Generic C = A(f32)[M,K] * W(f32)[K,N] + bias, WMMA f16 path -----------
// grid: (N/64, ceil(M/32)), block: 128 threads (4 waves; wave -> 16x32 tile)
__global__ __launch_bounds__(128) void gemm_f32_wmma(
    const float* __restrict__ A, long lda, const float* __restrict__ W,
    const float* __restrict__ bias, float* __restrict__ C, int M, int K, int N) {
  int w = threadIdx.x >> 5;
  int wm = w & 1, wn = w >> 1;
  int m0 = blockIdx.y * 32 + wm * 16;
  int n0 = blockIdx.x * 64 + wn * 32;
  if (m0 >= M) return;
  v8f acc0 = {}, acc1 = {};
  for (int k = 0; k < K; k += 32) {
    v16h a  = load_a_f32(A, lda, m0, k);
    v16h b0 = load_b_f32(W, N, k, n0);
    v16h b1 = load_b_f32(W, N, k, n0 + 16);
    acc0 = wmma_f16(a, b0, acc0);
    acc1 = wmma_f16(a, b1, acc1);
  }
  store_d_bias(C, N, m0, n0, acc0, bias);
  store_d_bias(C, N, m0, n0 + 16, acc1, bias);
}

// ---- Embedding gather ------------------------------------------------------
__global__ __launch_bounds__(256) void gather_emb(
    const float* __restrict__ table, const int* __restrict__ ids,
    float* __restrict__ emb, int total) {
  int i = blockIdx.x * 256 + threadIdx.x;
  if (i >= total) return;
  int row = i / kH, col = i - row * kH;
  emb[i] = table[(size_t)ids[row] * kH + col];
}

// ---- Fused GRU step: h_new = GRU(x_t, h_prev) ------------------------------
__global__ __launch_bounds__(128) void gru_step(
    const float* __restrict__ x, long ldx, const float* __restrict__ hprev,
    const float* __restrict__ Wih, const float* __restrict__ bih,
    const float* __restrict__ Whh, const float* __restrict__ bhh,
    float* __restrict__ hnew) {
  const int H = kH, N3 = 3 * kH;
  int w = threadIdx.x >> 5;
  int wm = w & 1, wn = w >> 1;
  int m0 = wm * 16;                        // batch rows (B=32 -> two 16-row tiles)
  int j0 = blockIdx.x * 32 + wn * 16;      // hidden column tile
  v8f gir = {}, giz = {}, gin = {}, ghr = {}, ghz = {}, ghn = {};
  for (int k = 0; k < H; k += 32) {
    v16h ax = load_a_f32(x, ldx, m0, k);
    v16h ah = load_a_f32(hprev, H, m0, k);
    v16h b0 = load_b_f32(Wih, N3, k, j0);
    v16h b1 = load_b_f32(Wih, N3, k, j0 + H);
    v16h b2 = load_b_f32(Wih, N3, k, j0 + 2 * H);
    gir = wmma_f16(ax, b0, gir);
    giz = wmma_f16(ax, b1, giz);
    gin = wmma_f16(ax, b2, gin);
    b0 = load_b_f32(Whh, N3, k, j0);
    b1 = load_b_f32(Whh, N3, k, j0 + H);
    b2 = load_b_f32(Whh, N3, k, j0 + 2 * H);
    ghr = wmma_f16(ah, b0, ghr);
    ghz = wmma_f16(ah, b1, ghz);
    ghn = wmma_f16(ah, b2, ghn);
  }
  int l = lane_id();
  int col = j0 + (l & 15);
  int mb = m0 + ((l >> 4) << 3);
  float bir = bih[col], biz = bih[col + H], bin = bih[col + 2 * H];
  float bhr = bhh[col], bhz = bhh[col + H], bhn = bhh[col + 2 * H];
#pragma unroll
  for (int r = 0; r < 8; ++r) {
    float rg = 1.0f / (1.0f + __expf(-(gir[r] + bir + ghr[r] + bhr)));
    float zg = 1.0f / (1.0f + __expf(-(giz[r] + biz + ghz[r] + bhz)));
    float ng = tanhf(gin[r] + bin + rg * (ghn[r] + bhn));
    float hp = hprev[(size_t)(mb + r) * H + col];
    hnew[(size_t)(mb + r) * H + col] = (1.0f - zg) * ng + zg * hp;
  }
}

// ---- Additive attention + softmax + context; writes combined row (f16) -----
__global__ __launch_bounds__(256) void attn_step(
    const float* __restrict__ qw, const float* __restrict__ vw,
    const float* __restrict__ values, const float* __restrict__ wc,
    const float* __restrict__ bc, const int* __restrict__ mask,
    const float* __restrict__ hnew, _Float16* __restrict__ combinedH, int t) {
  const int H = kH, S = kS;
  int b = blockIdx.x;
  __shared__ float sc[kS];
  int tid = threadIdx.x, wv = tid >> 5, l = tid & 31;
  const float* vwb = vw + (size_t)b * S * H;
  const float* qb = qw + (size_t)b * H;
  for (int s = wv * 16; s < wv * 16 + 16; ++s) {
    const float* row = vwb + (size_t)s * H;
    float acc = 0.0f;
    for (int h = l; h < H; h += 32) acc += wc[h] * tanhf(qb[h] + row[h]);
#pragma unroll
    for (int off = 16; off > 0; off >>= 1) acc += __shfl_down(acc, off, 32);
    if (l == 0) {
      float v = acc + bc[0];
      if (mask[b * S + s] == 0) v = -1e30f;
      sc[s] = v;
    }
  }
  __syncthreads();
  if (wv == 0) {  // softmax over S=128 in one wave
    float a0 = sc[l], a1 = sc[l + 32], a2 = sc[l + 64], a3 = sc[l + 96];
    float m = fmaxf(fmaxf(a0, a1), fmaxf(a2, a3));
#pragma unroll
    for (int off = 16; off > 0; off >>= 1) m = fmaxf(m, __shfl_xor(m, off, 32));
    float e0 = __expf(a0 - m), e1 = __expf(a1 - m), e2 = __expf(a2 - m), e3 = __expf(a3 - m);
    float ssum = e0 + e1 + e2 + e3;
#pragma unroll
    for (int off = 16; off > 0; off >>= 1) ssum += __shfl_xor(ssum, off, 32);
    float inv = 1.0f / ssum;
    sc[l] = e0 * inv; sc[l + 32] = e1 * inv; sc[l + 64] = e2 * inv; sc[l + 96] = e3 * inv;
  }
  __syncthreads();
  const float* valb = values + (size_t)b * S * H;
  _Float16* outrow = combinedH + ((size_t)b * kT + t) * (2 * H);
  for (int h = tid; h < H; h += 256) {
    float acc = 0.0f;
    for (int s = 0; s < S; ++s) acc += sc[s] * valb[(size_t)s * H + h];
    outrow[h] = (_Float16)hnew[(size_t)b * H + h];
    outrow[H + h] = (_Float16)acc;
  }
}

// ---- Logits GEMM: [B*T,2H](f16) x Wout[2H,V](f32) + bout -> f32 ------------
// Dominant GEMM (3072 x 2048 x 32000). Block 256 threads (8 waves),
// block tile 128M x 64N, wave tile 32x32 (4 accumulators, 4 WMMA / K-step).
// A K-slab (128x32 f16) DMA'd by the Tensor Data Mover into double-buffered
// LDS; Wout K-slab (32x64 f32) staged manually (coalesced loads + f16 convert).
__global__ __launch_bounds__(256) void logits_gemm(
    const _Float16* __restrict__ A, const float* __restrict__ W,
    const float* __restrict__ bias, float* __restrict__ C, int M, int K, int N) {
  __shared__ __align__(16) _Float16 As[2][128 * 32];  // [m][k], 64B rows
  __shared__ __align__(16) _Float16 Bs[2][64 * 48];   // [n][k], stride 48 halves
  int tid = threadIdx.x;
  int w = tid >> 5;
  int wm = w & 3, wn = w >> 2;
  int m0 = blockIdx.y * 128 + wm * 32;
  int nblk = blockIdx.x * 64;
  // staging assignment: thread -> (n = tid%64, k-group = tid/64 of 8 k's)
  int sn = tid & 63;
  int skg = tid >> 6;
  const float* g0 = W + (size_t)(skg * 8) * N + (nblk + sn);
  _Float16* dstB[2] = { &Bs[0][(size_t)sn * 48 + skg * 8],
                        &Bs[1][(size_t)sn * 48 + skg * 8] };
  const _Float16* arow = A + (size_t)(blockIdx.y * 128) * K;  // block's A rows
  unsigned asOff[2] = { (unsigned)(size_t)&As[0][0], (unsigned)(size_t)&As[1][0] };

  v8f acc00 = {}, acc01 = {}, acc10 = {}, acc11 = {};
  const int ksteps = K / 32;
  for (int kt = 0; kt < ksteps; ++kt) {
    int k0 = kt * 32;
    // TDM: DMA the 128x32 f16 A-slab into LDS (issued once, by wave 0)
    if (w == 0)
      tdm_load_2d_f16(asOff[kt & 1], arow + k0, /*tile_k=*/32, /*rows=*/128,
                      /*stride=*/(unsigned)K, /*tensor_d0=*/(unsigned)K,
                      /*tensor_d1=*/(unsigned)M);
    const float* g = g0 + (size_t)k0 * N;
    if (kt + 1 < ksteps)  // pull next K-slab of Wout toward L2 (global_prefetch_b8)
      __builtin_prefetch(g + (size_t)32 * N, 0, 0);
    // stage B: 8 coalesced f32 loads along K, convert, one 128-bit LDS store
    h8 hv;
#pragma unroll
    for (int j = 0; j < 8; ++j) hv[j] = (_Float16)g[(size_t)j * N];
    *reinterpret_cast<h8*>(dstB[kt & 1]) = hv;
    if (w == 0) __builtin_amdgcn_s_wait_tensorcnt(0);
    __syncthreads();   // releases all waves after wave0 saw TDM completion
    const _Float16* abuf = (kt & 1) ? &As[1][0] : &As[0][0];
    const _Float16* bbuf = (kt & 1) ? &Bs[1][0] : &Bs[0][0];
    v16h a0 = load_a_f16(abuf, 32, wm * 32, 0);
    v16h a1 = load_a_f16(abuf, 32, wm * 32 + 16, 0);
    v16h b0 = load_b_lds(bbuf, wn * 32);
    v16h b1 = load_b_lds(bbuf, wn * 32 + 16);
    acc00 = wmma_f16(a0, b0, acc00);
    acc01 = wmma_f16(a0, b1, acc01);
    acc10 = wmma_f16(a1, b0, acc10);
    acc11 = wmma_f16(a1, b1, acc11);
  }
  int n0 = nblk + wn * 32;
  store_d_bias(C, N, m0,      n0,      acc00, bias);
  store_d_bias(C, N, m0,      n0 + 16, acc01, bias);
  store_d_bias(C, N, m0 + 16, n0,      acc10, bias);
  store_d_bias(C, N, m0 + 16, n0 + 16, acc11, bias);
}

// ---------------------------------------------------------------------------
extern "C" void kernel_launch(void* const* d_in, const int* in_sizes, int n_in,
                              void* d_out, int out_size, void* d_ws, size_t ws_size,
                              hipStream_t stream) {
  const float* encoder_outputs = (const float*)d_in[0];   // [B,S,2H]
  const float* encoder_hidden  = (const float*)d_in[1];   // [B,2H]
  const int*   input_mask      = (const int*)d_in[2];     // [B,S]
  const int*   target_tensors  = (const int*)d_in[3];     // [B,T]
  const float* embedding       = (const float*)d_in[4];   // [V,H]
  const float* Wq   = (const float*)d_in[5];
  const float* bq   = (const float*)d_in[6];
  const float* Wv   = (const float*)d_in[7];
  const float* bv   = (const float*)d_in[8];
  const float* wc   = (const float*)d_in[9];
  const float* bc   = (const float*)d_in[10];
  const float* Wih  = (const float*)d_in[11];
  const float* bih  = (const float*)d_in[12];
  const float* Whh  = (const float*)d_in[13];
  const float* bhh  = (const float*)d_in[14];
  const float* Wproj = (const float*)d_in[15];
  const float* bproj = (const float*)d_in[16];
  const float* Wout  = (const float*)d_in[17];
  const float* bout  = (const float*)d_in[18];

  float* logits  = (float*)d_out;                       // [B,T,V]
  float* h_final = logits + (size_t)kB * kT * kV;       // [1,B,H]

  // Workspace carve-out (256B aligned), ~57 MB total.
  char* ws = (char*)d_ws;
  size_t off = 0;
  auto carve = [&](size_t bytes) -> void* {
    void* p = ws + off;
    off = (off + bytes + 255) & ~(size_t)255;
    return p;
  };
  float* emb    = (float*)carve((size_t)kB * kT * kH * 4);   // [B,T,H]
  float* values = (float*)carve((size_t)kB * kS * kH * 4);   // [B,S,H]
  float* vw     = (float*)carve((size_t)kB * kS * kH * 4);   // [B,S,H]
  float* hbuf0  = (float*)carve((size_t)kB * kH * 4);
  float* hbuf1  = (float*)carve((size_t)kB * kH * 4);
  float* qwbuf  = (float*)carve((size_t)kB * kH * 4);
  _Float16* combinedH = (_Float16*)carve((size_t)kB * kT * 2 * kH * 2);

  // --- setup ---
  {
    int total = kB * kT * kH;
    gather_emb<<<(total + 255) / 256, 256, 0, stream>>>(embedding, target_tensors, emb, total);
  }
  // h0 = encoder_hidden @ Wproj + bproj   [32,2048]x[2048,1024]
  gemm_f32_wmma<<<dim3(kH / 64, 1), 128, 0, stream>>>(
      encoder_hidden, 2 * kH, Wproj, bproj, hbuf0, kB, 2 * kH, kH);
  // values = encoder_outputs @ Wproj + bproj   [4096,2048]x[2048,1024]
  gemm_f32_wmma<<<dim3(kH / 64, (kB * kS) / 32), 128, 0, stream>>>(
      encoder_outputs, 2 * kH, Wproj, bproj, values, kB * kS, 2 * kH, kH);
  // vw = values @ Wv + bv   [4096,1024]x[1024,1024]
  gemm_f32_wmma<<<dim3(kH / 64, (kB * kS) / 32), 128, 0, stream>>>(
      values, kH, Wv, bv, vw, kB * kS, kH, kH);

  // --- recurrent steps (sequential dependency -> per-step launches) ---
  for (int t = 0; t < kT; ++t) {
    const float* hc = (t & 1) ? hbuf1 : hbuf0;
    float* hn       = (t & 1) ? hbuf0 : hbuf1;
    gru_step<<<kH / 32, 128, 0, stream>>>(
        emb + (size_t)t * kH, (long)kT * kH, hc, Wih, bih, Whh, bhh, hn);
    gemm_f32_wmma<<<dim3(kH / 64, 1), 128, 0, stream>>>(
        hn, kH, Wq, bq, qwbuf, kB, kH, kH);
    attn_step<<<kB, 256, 0, stream>>>(
        qwbuf, vw, values, wc, bc, input_mask, hn, combinedH, t);
  }

  // --- output projection (dominant GEMM): 3072x2048x32000 ---
  logits_gemm<<<dim3(kV / 64, (kB * kT) / 128), 256, 0, stream>>>(
      combinedH, Wout, bout, logits, kB * kT, 2 * kH, kV);

  // final hidden state: after 96 steps last write went to hbuf0
  (void)hipMemcpyAsync(h_final, hbuf0, (size_t)kB * kH * 4,
                       hipMemcpyDeviceToDevice, stream);
}
